// LSTMLayer_61375082660035
// MI455X (gfx1250) — matrix operand order repaired and checked by reading, not compiled, not run on hardware
//
#include <hip/hip_runtime.h>
#include <cstdint>

typedef __bf16 bf;
typedef __attribute__((ext_vector_type(16))) __bf16 v16bf;
typedef __attribute__((ext_vector_type(8)))  __bf16 v8bf;
typedef __attribute__((ext_vector_type(4)))  __bf16 v4bf;
typedef __attribute__((ext_vector_type(8)))  float  v8f;

#define BATCH 64
#define SEQT  2048
#define DIN   256
#define HID   128
#define NG    512            // 4*HID
#define BT    (BATCH*SEQT)   // 131072 rows

__device__ __forceinline__ bf f2bf(float x) {
  unsigned u = __builtin_bit_cast(unsigned, x);
  unsigned r = u + 0x7FFFu + ((u >> 16) & 1u);   // round-to-nearest-even
  unsigned short s = (unsigned short)(r >> 16);
  return __builtin_bit_cast(bf, s);
}

__device__ __forceinline__ v8f zero8() {
  v8f z = {0.f,0.f,0.f,0.f,0.f,0.f,0.f,0.f};
  return z;
}

__device__ __forceinline__ v8f wmma_bf16(v16bf a, v16bf b, v8f c) {
  return __builtin_amdgcn_wmma_f32_16x16x32_bf16(false, a, false, b, (short)0, c, false, false);
}

// Generic pointer to a __shared__ object: low 32 bits are the LDS offset
// (addrspacecast as3->flat places the SHARED_BASE aperture in bits [63:32]).
__device__ __forceinline__ unsigned lds_off_u32(const void* p) {
  return (unsigned)(unsigned long long)p;
}

// CDNA5 async global->LDS copy, 16B per lane, tracked by ASYNCcnt.
__device__ __forceinline__ void async_ld_b128(unsigned ldsoff, const void* g) {
  asm volatile("global_load_async_to_lds_b128 %0, %1, off"
               :: "v"(ldsoff), "v"(g) : "memory");
}
__device__ __forceinline__ void wait_async0() {
  asm volatile("s_wait_asynccnt 0x0" ::: "memory");
}

// A fragment 16x32 bf16 from row-major [M][K] storage.
// lane 0-15: M=lane, K {0..7,16..23}; lane 16-31: M=lane-16, K {8..15,24..31}
__device__ __forceinline__ v16bf load_a_frag(const bf* base, int stride, int kb) {
  int lane = threadIdx.x & 31;
  int r = lane & 15, hi = lane >> 4;
  const bf* p = base + r * stride + kb + hi * 8;
  union { v16bf v; v8bf h[2]; } u;
  u.h[0] = *(const v8bf*)p;
  u.h[1] = *(const v8bf*)(p + 16);
  return u.v;
}

// B fragment 32x16 bf16 of W^T from N-major [n][k] storage.
// lane 0-15: N=lane, K {0..7,16..23}; lane 16-31: N=lane-16, K {8..15,24..31}
__device__ __forceinline__ v16bf load_b_frag(const bf* base, int stride, int n0, int kb) {
  int lane = threadIdx.x & 31;
  int n = n0 + (lane & 15), hi = lane >> 4;
  const bf* p = base + n * stride + kb + hi * 8;
  union { v16bf v; v8bf h[2]; } u;
  u.h[0] = *(const v8bf*)p;
  u.h[1] = *(const v8bf*)(p + 16);
  return u.v;
}

// ---------------- small prep kernels ----------------
__global__ __launch_bounds__(256) void k_cvt_bf16(const float* __restrict__ s,
                                                  bf* __restrict__ d, int n) {
  int i = blockIdx.x * 256 + threadIdx.x;
  if (i < n) d[i] = f2bf(s[i]);
}

__global__ __launch_bounds__(256) void k_bias(const float* __restrict__ a,
                                              const float* __restrict__ b,
                                              float* __restrict__ d, int n) {
  int i = blockIdx.x * 256 + threadIdx.x;
  if (i < n) d[i] = a[i] + b[i];
}

// ---------------- bulk input GEMM: xw[M][512] = A[M][K] * W[512][K]^T + bias ----------------
#define LDT 40   // padded LDS row stride (halves): 80B, 16B-aligned rows

template <bool AF32>
__global__ __launch_bounds__(256) void lstm_xw_gemm(
    const float* __restrict__ Af32, const bf* __restrict__ Abf, int K,
    const bf* __restrict__ Bw, const float* __restrict__ bias,
    float* __restrict__ xw)
{
  __shared__ __attribute__((aligned(16))) bf at[128 * LDT];
  __shared__ __attribute__((aligned(16))) bf btl[128 * LDT];
  const int tid = threadIdx.x;
  const int m0 = (int)(blockIdx.x >> 2) * 128;
  const int n0 = (int)(blockIdx.x & 3) * 128;
  const int wid = tid >> 5, lane = tid & 31;
  const int wm = wid & 3, wn = wid >> 2;        // 4x2 wave grid over 128x128 tile
  const int nl = lane & 15, hi = lane >> 4;

  v8f acc[2][4];
  #pragma unroll
  for (int i = 0; i < 2; ++i)
    #pragma unroll
    for (int j = 0; j < 4; ++j) acc[i][j] = zero8();

  const int sr = tid >> 3;         // 0..31 (f32 A staging)
  const int sc = (tid & 7) * 4;    // 0..28

  for (int kt = 0; kt < K; kt += 32) {
    // B tile (bf16): async global->LDS, 512 x 16B chunks, 2 per thread
    #pragma unroll
    for (int k2 = 0; k2 < 2; ++k2) {
      int cc = tid + k2 * 256;
      int row = cc >> 2;
      int off = (cc & 3) * 8;      // halves
      async_ld_b128(lds_off_u32(&btl[row * LDT + off]),
                    Bw + (size_t)(n0 + row) * K + kt + off);
    }
    if (AF32) {
      // layer0 activations are f32: batch 4 loads, then convert+store
      float4 va[4];
      #pragma unroll
      for (int p = 0; p < 4; ++p)
        va[p] = *(const float4*)(Af32 + (size_t)(m0 + p * 32 + sr) * K + kt + sc);
      #pragma unroll
      for (int p = 0; p < 4; ++p) {
        v4bf t; t[0] = f2bf(va[p].x); t[1] = f2bf(va[p].y);
        t[2] = f2bf(va[p].z); t[3] = f2bf(va[p].w);
        *(v4bf*)(at + (p * 32 + sr) * LDT + sc) = t;
      }
    } else {
      // layer1 activations are bf16: async direct to LDS
      #pragma unroll
      for (int k2 = 0; k2 < 2; ++k2) {
        int cc = tid + k2 * 256;
        int row = cc >> 2;
        int off = (cc & 3) * 8;
        async_ld_b128(lds_off_u32(&at[row * LDT + off]),
                      Abf + (size_t)(m0 + row) * K + kt + off);
      }
    }
    wait_async0();
    __syncthreads();

    // hoist all fragment loads, then issue the 8 WMMAs back-to-back
    v16bf a0 = load_a_frag(at + (wm * 32 +  0) * LDT, LDT, 0);
    v16bf a1 = load_a_frag(at + (wm * 32 + 16) * LDT, LDT, 0);
    v16bf bb[4];
    #pragma unroll
    for (int fn = 0; fn < 4; ++fn)
      bb[fn] = load_b_frag(btl, LDT, wn * 64 + fn * 16, 0);
    #pragma unroll
    for (int fn = 0; fn < 4; ++fn) {
      acc[0][fn] = wmma_bf16(a0, bb[fn], acc[0][fn]);
      acc[1][fn] = wmma_bf16(a1, bb[fn], acc[1][fn]);
    }
    __syncthreads();
  }

  #pragma unroll
  for (int fm = 0; fm < 2; ++fm)
    #pragma unroll
    for (int fn = 0; fn < 4; ++fn) {
      int gc = n0 + wn * 64 + fn * 16 + nl;
      float bv = bias[gc];
      size_t rb = (size_t)(m0 + wm * 32 + fm * 16 + 8 * hi);
      #pragma unroll
      for (int v = 0; v < 8; ++v)
        xw[(rb + v) * NG + gc] = acc[fm][fn][v] + bv;
    }
}

// ---------------- persistent recurrent scan ----------------
__device__ __forceinline__ float sigm(float x) { return 1.f / (1.f + __expf(-x)); }
__device__ __forceinline__ float tanh_fast(float x) {
  x = fminf(fmaxf(x, -20.f), 20.f);
  float e = __expf(-2.f * x);
  return (1.f - e) / (1.f + e);
}

#define GST 520   // gates row stride (f32)
#define HST 136   // h row stride (bf16)

// 4 blocks x 512 threads (16 waves). Block bg owns batch rows [bg*16, bg*16+16).
// Whh fragments live in registers (loop invariant); h in LDS; xw(t+1) streams
// into a double-buffered LDS gates tile via async global->LDS during step t.
__global__ __launch_bounds__(512) void lstm_scan(
    const float* __restrict__ xw, const bf* __restrict__ Whh,
    float* __restrict__ outF, bf* __restrict__ outB)
{
  __shared__ __attribute__((aligned(16))) float gbuf[2][16 * GST];
  __shared__ __attribute__((aligned(16))) bf hlds[16 * HST];
  const int tid = threadIdx.x;
  const int bg  = blockIdx.x;
  const int wid = tid >> 5, lane = tid & 31;
  const int nl = lane & 15, hi = lane >> 4;
  const int nbase = wid * 32;   // each of 16 waves owns 32 gate columns

  // loop-invariant B fragments of Whh (natural [512][128] bf16 layout)
  v16bf bfrag[4][2];
  #pragma unroll
  for (int kb = 0; kb < 4; ++kb)
    #pragma unroll
    for (int fn = 0; fn < 2; ++fn)
      bfrag[kb][fn] = load_b_frag(Whh, HID, nbase + fn * 16, kb * 32);

  for (int i = tid; i < 16 * HST; i += 512)
    hlds[i] = __builtin_bit_cast(bf, (unsigned short)0);

  const int eb = tid >> 5;          // elementwise: batch row
  const int eh = (tid & 31) * 4;    // elementwise: 4 hidden cols
  float c[4] = {0.f, 0.f, 0.f, 0.f};

  const int pb = tid >> 5;          // staging: row 0..15
  const int pc = (tid & 31) * 16;   // staging: 16 gate cols (64B = 4 x b128)
  const float* xwbase = xw + (size_t)(bg * 16 + pb) * SEQT * NG + pc;
  unsigned loff0 = lds_off_u32(&gbuf[0][pb * GST + pc]);
  unsigned loff1 = lds_off_u32(&gbuf[1][pb * GST + pc]);

  // kick off t=0
  #pragma unroll
  for (int j = 0; j < 4; ++j)
    async_ld_b128(loff0 + j * 16, xwbase + j * 4);

  for (int t = 0; t < SEQT; ++t) {
    float* gates = gbuf[t & 1];
    wait_async0();        // this wave's xw(t) chunks have landed in LDS
    __syncthreads();      // everyone's chunks visible; prev elementwise done
    if (t + 1 < SEQT) {   // stream xw(t+1) into the other buffer during compute
      const float* src = xwbase + (size_t)(t + 1) * NG;
      unsigned lo = ((t + 1) & 1) ? loff1 : loff0;
      #pragma unroll
      for (int j = 0; j < 4; ++j)
        async_ld_b128(lo + j * 16, src + j * 4);
    }

    // gates[16 x 32 slice] += h(16x128) * Whh^T(128x32 slice)
    // hoist all 4 A fragments so the 8 ds_load_b128 issue back-to-back,
    // then the 8 WMMAs drain them with staggered dscnt waits
    v16bf a[4];
    #pragma unroll
    for (int kb = 0; kb < 4; ++kb)
      a[kb] = load_a_frag(hlds, HST, kb * 32);
    v8f acc[2];
    acc[0] = zero8(); acc[1] = zero8();
    #pragma unroll
    for (int kb = 0; kb < 4; ++kb) {
      acc[0] = wmma_bf16(a[kb], bfrag[kb][0], acc[0]);
      acc[1] = wmma_bf16(a[kb], bfrag[kb][1], acc[1]);
    }
    #pragma unroll
    for (int fn = 0; fn < 2; ++fn) {
      int col = nbase + fn * 16 + nl;
      #pragma unroll
      for (int v = 0; v < 8; ++v)
        gates[(v + 8 * hi) * GST + col] += acc[fn][v];
    }
    __syncthreads();

    // elementwise LSTM cell (gate order i,f,g,o), c in registers
    float hv[4];
    #pragma unroll
    for (int j = 0; j < 4; ++j) {
      int hc = eh + j;
      float ig = sigm(gates[eb * GST + hc]);
      float fg = sigm(gates[eb * GST + HID + hc]);
      float gg = tanh_fast(gates[eb * GST + 2 * HID + hc]);
      float og = sigm(gates[eb * GST + 3 * HID + hc]);
      c[j] = fg * c[j] + ig * gg;
      hv[j] = og * tanh_fast(c[j]);
      hlds[eb * HST + hc] = f2bf(hv[j]);
    }
    size_t orow = ((size_t)(bg * 16 + eb) * SEQT + t) * (size_t)HID + eh;
    if (outF) {
      *(float4*)(outF + orow) = make_float4(hv[0], hv[1], hv[2], hv[3]);
    } else {
      union { bf a4[4]; uint2 u; } pk;
      #pragma unroll
      for (int j = 0; j < 4; ++j) pk.a4[j] = f2bf(hv[j]);
      *(uint2*)(outB + orow) = pk.u;
    }
    __syncthreads();
  }
}

// ---------------- launch ----------------
extern "C" void kernel_launch(void* const* d_in, const int* in_sizes, int n_in,
                              void* d_out, int out_size, void* d_ws, size_t ws_size,
                              hipStream_t stream) {
  (void)in_sizes; (void)n_in; (void)out_size; (void)ws_size;
  const float* x    = (const float*)d_in[0];
  const float* Wih0 = (const float*)d_in[1];
  const float* Whh0 = (const float*)d_in[2];
  const float* bih0 = (const float*)d_in[3];
  const float* bhh0 = (const float*)d_in[4];
  const float* Wih1 = (const float*)d_in[5];
  const float* Whh1 = (const float*)d_in[6];
  const float* bih1 = (const float*)d_in[7];
  const float* bhh1 = (const float*)d_in[8];

  char* ws = (char*)d_ws;
  float* xwbuf = (float*)(ws);                      // 131072*512*4 = 268435456 B (reused by both layers)
  bf*    h1b   = (bf*)(ws + 268435456ull);          // 131072*128*2 =  33554432 B
  bf*    wih0b = (bf*)(ws + 301989888ull);          // 512*256*2
  bf*    whh0b = (bf*)(ws + 302252032ull);          // 512*128*2
  bf*    wih1b = (bf*)(ws + 302383104ull);          // 512*128*2
  bf*    whh1b = (bf*)(ws + 302514176ull);          // 512*128*2
  float* b0    = (float*)(ws + 302645248ull);       // 512*4
  float* b1    = (float*)(ws + 302647296ull);       // 512*4

  // weight conversion + fused biases
  k_cvt_bf16<<<512, 256, 0, stream>>>(Wih0, wih0b, NG * DIN);
  k_cvt_bf16<<<256, 256, 0, stream>>>(Whh0, whh0b, NG * HID);
  k_cvt_bf16<<<256, 256, 0, stream>>>(Wih1, wih1b, NG * HID);
  k_cvt_bf16<<<256, 256, 0, stream>>>(Whh1, whh1b, NG * HID);
  k_bias<<<2, 256, 0, stream>>>(bih0, bhh0, b0, NG);
  k_bias<<<2, 256, 0, stream>>>(bih1, bhh1, b1, NG);

  // layer 0
  lstm_xw_gemm<true><<<dim3((BT / 128) * 4), dim3(256), 0, stream>>>(x, nullptr, DIN, wih0b, b0, xwbuf);
  lstm_scan<<<dim3(4), dim3(512), 0, stream>>>(xwbuf, whh0b, nullptr, h1b);

  // layer 1 (xw buffer reused; kernels serialize on the stream)
  lstm_xw_gemm<false><<<dim3((BT / 128) * 4), dim3(256), 0, stream>>>(nullptr, h1b, HID, wih1b, b1, xwbuf);
  lstm_scan<<<dim3(4), dim3(512), 0, stream>>>(xwbuf, whh1b, (float*)d_out, nullptr);
}
